// GQA_8632884265311
// MI455X (gfx1250) — matrix operand order repaired and checked by reading, not compiled, not run on hardware
//
#include <hip/hip_runtime.h>
#include <hip/hip_bf16.h>

typedef __attribute__((ext_vector_type(16))) _Float16     v16h;
typedef __attribute__((ext_vector_type(8)))  float        v8f;
typedef __attribute__((ext_vector_type(4)))  unsigned int u32x4;
typedef __attribute__((ext_vector_type(8)))  int          i32x8;
typedef __attribute__((ext_vector_type(4)))  int          i32x4;

union Frag16 {
    v16h h;
    unsigned int u[8];
    _Float16 f[16];
};

// A-fragment K index, contiguous K order (matches B fragments):
// lane group g = lane>>4; VGPR v holds the K pair starting at:
__device__ __forceinline__ int a_k_of(int v, int g) {
    return 16 * g + 2 * v;
}

// ---------------------------------------------------------------------------
// WMMA GEMM: C[M,N] = A[M,K] * B[K,N]
// A: AT (float or _Float16) row-major; B: float row-major; C: OT.
// Block tile 128x128, 256 threads (8 waves), wave tile 32x64, K-step 32.
// Software-pipelined: tile k+1 loaded to registers while tile k computes,
// then converted/stored into the alternate LDS buffer (1 barrier/iter).
// TOUT: store output transposed as [row>>11][col][row&2047] (for V scratch).
// ---------------------------------------------------------------------------
template <typename AT, typename OT, bool TOUT>
__global__ __launch_bounds__(256) void gemm_wmma_f16(
    const AT* __restrict__ A, const float* __restrict__ B, OT* __restrict__ C,
    int M, int N, int K)
{
    __shared__ _Float16 As[2][128][34];   // +2 pad: conflict-free A-frag reads
    __shared__ _Float16 Bt[2][128][34];   // B transposed: Bt[n][k], +2 pad

    const int tid  = threadIdx.x;
    const int lane = tid & 31;
    const int wid  = tid >> 5;
    const int wr   = wid & 3;          // wave row -> m offset 32*wr
    const int wc   = wid >> 2;         // wave col -> n offset 64*wc
    const int g    = lane >> 4;
    const int ln   = lane & 15;
    const long m0  = (long)blockIdx.y * 128;
    const long n0  = (long)blockIdx.x * 128;

    float ar[16], br[16];              // register staging for next tile

    auto loadT = [&](int k0) {
#pragma unroll
        for (int j = 0; j < 16; ++j) {
            int i = tid + 256 * j;     // A tile 128x32
            ar[j] = (float)A[(m0 + (i >> 5)) * (long)K + k0 + (i & 31)];
        }
#pragma unroll
        for (int j = 0; j < 16; ++j) {
            int i = tid + 256 * j;     // B tile 32x128
            br[j] = B[(long)(k0 + (i >> 7)) * N + n0 + (i & 127)];
        }
    };
    auto storeT = [&](int buf) {
#pragma unroll
        for (int j = 0; j < 16; ++j) {
            int i = tid + 256 * j;
            As[buf][i >> 5][i & 31] = (_Float16)ar[j];
        }
#pragma unroll
        for (int j = 0; j < 16; ++j) {
            int i = tid + 256 * j;
            Bt[buf][i & 127][i >> 7] = (_Float16)br[j];
        }
    };

    v8f acc[2][4] = {};

    loadT(0);
    storeT(0);
    __syncthreads();

    for (int k0 = 0; k0 < K; k0 += 32) {
        const int cur  = (k0 >> 5) & 1;
        const bool more = (k0 + 32) < K;
        // Prefetch 2 tiles ahead (global_prefetch_b8)
        if (k0 + 64 < K) {
            __builtin_prefetch(A + (m0 + (tid & 127)) * (long)K + (k0 + 64), 0, 3);
            if (tid < 32)
                __builtin_prefetch(B + (long)(k0 + 64 + tid) * N + n0, 0, 3);
        }
        if (more) loadT(k0 + 32);      // global loads overlap the WMMAs below

        Frag16 a[2], b[4];
#pragma unroll
        for (int mt = 0; mt < 2; ++mt)
#pragma unroll
            for (int v = 0; v < 8; ++v)
                a[mt].u[v] = *(const unsigned int*)&As[cur][32 * wr + 16 * mt + ln][a_k_of(v, g)];
#pragma unroll
        for (int nt = 0; nt < 4; ++nt)
#pragma unroll
            for (int v = 0; v < 8; ++v)
                b[nt].u[v] = *(const unsigned int*)&Bt[cur][64 * wc + 16 * nt + ln][16 * g + 2 * v];

#pragma unroll
        for (int nt = 0; nt < 4; ++nt)
#pragma unroll
            for (int mt = 0; mt < 2; ++mt)
                acc[mt][nt] = __builtin_amdgcn_wmma_f32_16x16x32_f16(
                    false, a[mt].h, false, b[nt].h, (short)0, acc[mt][nt], false, false);

        if (more) storeT(1 - cur);     // fill alternate buffer
        __syncthreads();
    }

    // C layout: VGPR r, lane -> row r+8g, col ln
#pragma unroll
    for (int mt = 0; mt < 2; ++mt)
#pragma unroll
        for (int nt = 0; nt < 4; ++nt)
#pragma unroll
            for (int r = 0; r < 8; ++r) {
                long row = m0 + 32 * wr + 16 * mt + r + 8 * g;
                long col = n0 + 64 * wc + 16 * nt + ln;
                if constexpr (TOUT)
                    C[((row >> 11) * (long)N + col) * 2048 + (row & 2047)] =
                        (OT)acc[mt][nt][r];
                else
                    C[row * (long)N + col] = (OT)acc[mt][nt][r];
            }
}

// ---------------------------------------------------------------------------
// Flash-attention (causal, cur_pos=0), GQA group = h/4.
// Q:  [B*S, 4096] f16 (col = h*128+d)
// Kc: [B*S, 1024] f16 (col = g*128+d), natural layout
// Vt: [B*G, 128, S] f16, pre-transposed by the V GEMM (TDM-friendly rows)
// O:  [B*S, 4096] f16. Block = (b, h, 128 q rows), 8 waves x 16 rows.
// K/V tiles double-buffered via the Tensor Data Mover: wave 0 issues tile
// kt+1 while all waves compute tile kt; s_wait_tensorcnt(2) exploits the
// per-wave in-order completion of tensor ops. TDM LDS padding gives
// bank-conflict-free fragment reads.
// ---------------------------------------------------------------------------
__global__ __launch_bounds__(256) void attn_wmma_f16(
    const _Float16* __restrict__ Q, const _Float16* __restrict__ Kc,
    const _Float16* __restrict__ Vtr, _Float16* __restrict__ O)
{
    __shared__ _Float16 Ks[2][32][136];    // [key][hd], +8 pad (272B rows)
    __shared__ _Float16 Vt[2][128][40];    // [hd][key], +8 pad (80B rows)
    __shared__ _Float16 Ps[8][16][34];     // per-wave P bounce buffer

    const int tid  = threadIdx.x;
    const int lane = tid & 31;
    const int wid  = tid >> 5;
    const int g    = lane >> 4;
    const int ln   = lane & 15;

    const int q0  = blockIdx.x * 128;
    const int h   = blockIdx.y;
    const int b   = blockIdx.z;
    const int grp = h >> 2;                // repeat_interleave: head -> group h/4
    const long rb = (long)b * 2048;        // batch row base

    // Issue TDM loads of one K tile + one V tile into buffer `buf`.
    auto tdm_issue = [&](int kbase, int buf) {
        i32x4 zz4 = {0, 0, 0, 0};
        i32x8 zz8 = {0, 0, 0, 0, 0, 0, 0, 0};

        // K tile: 32 rows x 128 f16, row stride 1024; LDS pad 4dw/64dw -> 272B rows
        unsigned long long ka =
            (unsigned long long)(const void*)(Kc + (rb + kbase) * 1024 + (long)grp * 128);
        u32x4 kg0;
        kg0[0] = 1u;                                   // count=1
        kg0[1] = (unsigned)(unsigned long long)(void*)&Ks[buf][0][0];
        kg0[2] = (unsigned)ka;                         // global_addr lo
        kg0[3] = (unsigned)(ka >> 32) | 0x80000000u;   // addr hi | type=2
        i32x8 kg1;
        kg1[0] = (1 << 16) | (1 << 20) | (5 << 22) | (3 << 25); // 2B, pad 4dw/64dw
        kg1[1] = (128 << 16);                          // tensor_dim0 = 128
        kg1[2] = (int)0xFFFF0000u;                     // tensor_dim1 = 65535
        kg1[3] = (128 << 16);                          // tile_dim0 = 128
        kg1[4] = 32;                                   // tile_dim1 = 32
        kg1[5] = 1024;                                 // dim0 stride = 1024
        kg1[6] = 0;
        kg1[7] = 0;
        __builtin_amdgcn_tensor_load_to_lds(kg0, kg1, zz4, zz4, zz8, 0);

        // V tile: 128 rows x 32 f16 (pre-transposed), row stride 2048;
        // LDS pad 4dw/16dw -> 80B rows
        unsigned long long va = (unsigned long long)(const void*)
            (Vtr + ((long)(b * 8 + grp) * 128) * 2048 + kbase);
        u32x4 vg0;
        vg0[0] = 1u;
        vg0[1] = (unsigned)(unsigned long long)(void*)&Vt[buf][0][0];
        vg0[2] = (unsigned)va;
        vg0[3] = (unsigned)(va >> 32) | 0x80000000u;
        i32x8 vg1;
        vg1[0] = (1 << 16) | (1 << 20) | (3 << 22) | (3 << 25); // 2B, pad 4dw/16dw
        vg1[1] = (32 << 16);                           // tensor_dim0 = 32
        vg1[2] = (int)0xFFFF0000u;                     // tensor_dim1 = 65535
        vg1[3] = (32 << 16);                           // tile_dim0 = 32
        vg1[4] = 128;                                  // tile_dim1 = 128
        vg1[5] = 2048;                                 // dim0 stride = 2048
        vg1[6] = 0;
        vg1[7] = 0;
        __builtin_amdgcn_tensor_load_to_lds(vg0, vg1, zz4, zz4, zz8, 0);
    };

    // Preload Q fragments for this wave's 16 rows (hd = 4 chunks of 32)
    Frag16 qf[4];
    {
        const _Float16* qp = Q + (rb + q0 + 16 * wid + ln) * 4096 + (long)h * 128;
#pragma unroll
        for (int ks = 0; ks < 4; ++ks)
#pragma unroll
            for (int v = 0; v < 8; ++v)
                qf[ks].u[v] = *(const unsigned int*)(qp + ks * 32 + a_k_of(v, g));
    }

    v8f oacc[8] = {};
    float mrow[8], lrow[8];
#pragma unroll
    for (int r = 0; r < 8; ++r) { mrow[r] = -1e30f; lrow[r] = 0.f; }

    const float scale = 0.08838834764831845f;   // 1/sqrt(128)
    const int nkt = (q0 + 128) / 32;            // causal upper bound for this block

    // Prologue: DMA tile 0 into buffer 0
    if (__builtin_amdgcn_readfirstlane(wid) == 0)
        tdm_issue(0, 0);

    for (int kt = 0; kt < nkt; ++kt) {
        const int cur   = kt & 1;
        const int kbase = kt * 32;

        // Pipeline: issue tile kt+1 into the other buffer, then wait for
        // tile kt (tensor ops complete in order; 2 newest may be pending).
        if (__builtin_amdgcn_readfirstlane(wid) == 0) {
            if (kt + 1 < nkt) {
                tdm_issue(kbase + 32, 1 - cur);
                __builtin_amdgcn_s_wait_tensorcnt(2);
            } else {
                __builtin_amdgcn_s_wait_tensorcnt(0);
            }
        }
        __syncthreads();   // buf[cur] ready for all waves

        // Scores: 16 rows x 32 keys = two 16x16 accumulators, K-dim = hd (4x32)
        v8f sc[2];
#pragma unroll
        for (int t = 0; t < 2; ++t) {
            Frag16 bf[4];  // B[kk][n] = K[key=16t+n][hd=ks*32+kk]
#pragma unroll
            for (int ks = 0; ks < 4; ++ks)
#pragma unroll
                for (int v = 0; v < 8; ++v)
                    bf[ks].u[v] =
                        *(const unsigned int*)&Ks[cur][16 * t + ln][ks * 32 + 16 * g + 2 * v];
            v8f s = {};
#pragma unroll
            for (int ks = 0; ks < 4; ++ks)
                s = __builtin_amdgcn_wmma_f32_16x16x32_f16(
                    false, qf[ks].h, false, bf[ks].h, (short)0, s, false, false);
            sc[t] = s;
        }

        // Causal mask + online softmax (row r+8g lives across 16 lanes)
        const int rowb = q0 + 16 * wid;
        float tmax[8];
#pragma unroll
        for (int r = 0; r < 8; ++r) {
            int qrow = rowb + r + 8 * g;
#pragma unroll
            for (int t = 0; t < 2; ++t) {
                int key = kbase + 16 * t + ln;
                float x = sc[t][r] * scale;
                sc[t][r] = (key <= qrow) ? x : -1e30f;
            }
            float mx = fmaxf(sc[0][r], sc[1][r]);
#pragma unroll
            for (int off = 1; off < 16; off <<= 1)
                mx = fmaxf(mx, __shfl_xor(mx, off, 32));
            tmax[r] = mx;
        }

        float alpha[8];
#pragma unroll
        for (int r = 0; r < 8; ++r) {
            float mnew = fmaxf(mrow[r], tmax[r]);
            alpha[r] = __expf(mrow[r] - mnew);
            mrow[r] = mnew;
            float p0 = __expf(sc[0][r] - mnew);
            float p1 = __expf(sc[1][r] - mnew);
            sc[0][r] = p0; sc[1][r] = p1;
            float ls = p0 + p1;
#pragma unroll
            for (int off = 1; off < 16; off <<= 1)
                ls += __shfl_xor(ls, off, 32);
            lrow[r] = lrow[r] * alpha[r] + ls;
#pragma unroll
            for (int nt = 0; nt < 8; ++nt)
                oacc[nt][r] *= alpha[r];
        }

        // P: C-layout -> A-fragment layout via per-wave LDS (wave LDS in-order)
#pragma unroll
        for (int t = 0; t < 2; ++t)
#pragma unroll
            for (int r = 0; r < 8; ++r)
                Ps[wid][r + 8 * g][16 * t + ln] = (_Float16)sc[t][r];

        Frag16 pf;
#pragma unroll
        for (int v = 0; v < 8; ++v)
            pf.u[v] = *(const unsigned int*)&Ps[wid][ln][a_k_of(v, g)];

        // PV: out[16 x 128] += P[16 x 32keys] * V[32keys x 128]
#pragma unroll
        for (int nt = 0; nt < 8; nt += 2) {
            Frag16 vf[2];  // B[kk][n] = V[key=16g+2v(+1)][hd=16(nt+p)+n]
#pragma unroll
            for (int p = 0; p < 2; ++p)
#pragma unroll
                for (int v = 0; v < 8; ++v)
                    vf[p].u[v] =
                        *(const unsigned int*)&Vt[cur][16 * (nt + p) + ln][16 * g + 2 * v];
#pragma unroll
            for (int p = 0; p < 2; ++p)
                oacc[nt + p] = __builtin_amdgcn_wmma_f32_16x16x32_f16(
                    false, pf.h, false, vf[p].h, (short)0, oacc[nt + p], false, false);
        }

        __syncthreads();   // all waves done reading buf[cur]; DMA may overwrite
    }

    // Normalize and store [B,S,H,hd] -> O[b*S+s][h*128 + d]
    float inv[8];
#pragma unroll
    for (int r = 0; r < 8; ++r) inv[r] = 1.0f / lrow[r];
#pragma unroll
    for (int nt = 0; nt < 8; ++nt)
#pragma unroll
        for (int r = 0; r < 8; ++r) {
            long s = q0 + 16 * wid + r + 8 * g;
            O[(rb + s) * 4096 + (long)h * 128 + 16 * nt + ln] =
                (_Float16)(oacc[nt][r] * inv[r]);
        }
}

// ---------------------------------------------------------------------------
extern "C" void kernel_launch(void* const* d_in, const int* in_sizes, int n_in,
                              void* d_out, int out_size, void* d_ws, size_t ws_size,
                              hipStream_t stream) {
    (void)in_sizes; (void)n_in; (void)out_size; (void)ws_size;
    const float* x  = (const float*)d_in[0];   // [2,2048,4096]
    const float* wq = (const float*)d_in[1];   // [4096,4096]
    const float* wk = (const float*)d_in[2];   // [4096,1024]
    const float* wv = (const float*)d_in[3];   // [4096,1024]
    const float* wo = (const float*)d_in[4];   // [4096,4096]
    // d_in[5] = cur_pos (0 in reference; standard causal mask)

    const size_t M  = 4096;                    // B*S
    const size_t qE = M * 4096;
    const size_t kE = M * 1024;
    _Float16* qs = (_Float16*)d_ws;            // 32 MB
    _Float16* ks = qs + qE;                    //  8 MB
    _Float16* vs = ks + kE;                    //  8 MB, TRANSPOSED [b][g*128+d][S]
    _Float16* ao = vs + kE;                    // 32 MB  (total 80 MB)

    dim3 blk(256);
    // QKV projections (fp32 in -> f16 out); V written pre-transposed for TDM.
    gemm_wmma_f16<float, _Float16, false><<<dim3(32, 32), blk, 0, stream>>>(
        x, wq, qs, 4096, 4096, 4096);
    gemm_wmma_f16<float, _Float16, false><<<dim3(8, 32), blk, 0, stream>>>(
        x, wk, ks, 4096, 1024, 4096);
    gemm_wmma_f16<float, _Float16, true><<<dim3(8, 32), blk, 0, stream>>>(
        x, wv, vs, 4096, 1024, 4096);
    // Flash attention: (q_tiles=16, heads=32, batch=2)
    attn_wmma_f16<<<dim3(16, 32, 2), blk, 0, stream>>>(qs, ks, vs, ao);
    // Output projection (f16 in -> fp32 out)
    gemm_wmma_f16<_Float16, float, false><<<dim3(32, 32), blk, 0, stream>>>(
        ao, wo, (float*)d_out, 4096, 4096, 4096);
}
